// Diffusion_Model_14010183319916
// MI455X (gfx1250) — compile-verified, matrix operand order checked by reading, not run on hardware
//
#include <hip/hip_runtime.h>
#include <hip/hip_bf16.h>
#include <float.h>

typedef __attribute__((ext_vector_type(16))) _Float16 v16h;
typedef __attribute__((ext_vector_type(8)))  float    v8f;

#define BC 64
#define TT 32

// ---------------------------------------------------------------------------
// helpers
// ---------------------------------------------------------------------------

// Load 16 f16 elements for a WMMA A/B fragment from an f32 row `p` (32 floats):
// frag[0..7]  = p[k0 .. k0+7]
// frag[8..15] = p[k0+16 .. k0+23]
// (k0 = 0 for lanes 0-15, 8 for lanes 16-31 — per ISA 16-bit 16x32 layout)
__device__ __forceinline__ v16h frag_load_f32(const float* p, int k0) {
  const float4 f0 = *(const float4*)(p + k0);
  const float4 f1 = *(const float4*)(p + k0 + 4);
  const float4 f2 = *(const float4*)(p + k0 + 16);
  const float4 f3 = *(const float4*)(p + k0 + 20);
  v16h a;
  a[0]  = (_Float16)f0.x; a[1]  = (_Float16)f0.y; a[2]  = (_Float16)f0.z; a[3]  = (_Float16)f0.w;
  a[4]  = (_Float16)f1.x; a[5]  = (_Float16)f1.y; a[6]  = (_Float16)f1.z; a[7]  = (_Float16)f1.w;
  a[8]  = (_Float16)f2.x; a[9]  = (_Float16)f2.y; a[10] = (_Float16)f2.z; a[11] = (_Float16)f2.w;
  a[12] = (_Float16)f3.x; a[13] = (_Float16)f3.y; a[14] = (_Float16)f3.z; a[15] = (_Float16)f3.w;
  return a;
}

// DPP ROW_XMASK butterfly add: x += lane(x ^ MASK) within a 16-lane row.
// DPP16 ctrl = 0x160 | mask (ROW_XMASK), all rows/banks enabled.
// (compiler fuses into a single v_add_f32_dpp)
template <int CTRL>
__device__ __forceinline__ float dpp_xadd(float x) {
  int xi = __float_as_int(x);
  int yi = __builtin_amdgcn_update_dpp(xi, xi, CTRL, 0xF, 0xF, true);
  return x + __int_as_float(yi);
}

// sum across a 16-lane half-group (wave32) — pure VALU, no LDS traffic
__device__ __forceinline__ float red16(float x) {
  x = dpp_xadd<0x161>(x);   // xor 1
  x = dpp_xadd<0x162>(x);   // xor 2
  x = dpp_xadd<0x164>(x);   // xor 4
  x = dpp_xadd<0x168>(x);   // xor 8
  return x;
}

__device__ __forceinline__ void atomicMaxFloat(float* addr, float val) {
  unsigned int* ua = (unsigned int*)addr;
  unsigned int old = __float_as_uint(*addr);
  while (val > __uint_as_float(old)) {
    unsigned int assumed = old;
    old = atomicCAS(ua, assumed, __float_as_uint(val));
    if (old == assumed) break;
  }
}

// One NAM branch for a 16-row tile:
//   sigmoid( relu( LN( A @ Wt + b ) ) @ w12 + bout )
// A: 16x32 f16 frag.  b0/b1: W-columns h=0..15 / 16..31 as B frags.
// Per-lane constants indexed by h0 = lane&15, h1 = 16+(lane&15).
// Result: out8[v] = sigmoid value for row M = v + (lane<16 ? 0 : 8),
// replicated across each 16-lane group.
__device__ __forceinline__ void nam_branch(
    const v16h a, const v16h b0, const v16h b1,
    float bb0, float bb1, float g0, float g1, float e0, float e1,
    float w0, float w1, float bout, float* out8)
{
  v8f c = {};
  v8f d0 = __builtin_amdgcn_wmma_f32_16x16x32_f16(false, a, false, b0, (short)0, c, false, false);
  v8f d1 = __builtin_amdgcn_wmma_f32_16x16x32_f16(false, a, false, b1, (short)0, c, false, false);
#pragma unroll
  for (int v = 0; v < 8; ++v) {
    float x0 = d0[v] + bb0;          // h = lane&15
    float x1 = d1[v] + bb1;          // h = 16 + (lane&15)
    float s  = red16(x0 + x1);       // sum over 32 h-values of this row
    float ss = red16(x0*x0 + x1*x1);
    float mu   = s * (1.0f / 32.0f);
    float var  = ss * (1.0f / 32.0f) - mu * mu;
    float rstd = rsqrtf(var + 1e-5f);
    float y0 = (x0 - mu) * rstd * g0 + e0;
    float y1 = (x1 - mu) * rstd * g1 + e1;
    y0 = fmaxf(y0, 0.0f);
    y1 = fmaxf(y1, 0.0f);
    float dot = red16(y0 * w0 + y1 * w1);
    out8[v] = 1.0f / (1.0f + __expf(-(dot + bout)));
  }
}

// ---------------------------------------------------------------------------
// kernels
// ---------------------------------------------------------------------------

// fold attn_w into fc_w:  w_src[t] = sum_h attn[h]*fc[h,t];  w_dst uses attn[H+h]
__global__ void k_fold(const float* __restrict__ fc_w, const float* __restrict__ attn_w,
                       float* __restrict__ wsrc, float* __restrict__ wdst) {
  int t = threadIdx.x;
  if (t >= TT) return;
  float a = 0.f, d = 0.f;
  for (int h = 0; h < 32; ++h) {
    float f = fc_w[h * TT + t];
    a += attn_w[h] * f;
    d += attn_w[32 + h] * f;
  }
  wsrc[t] = a;
  wdst[t] = d;
}

__global__ void k_init(float* __restrict__ pred, float* __restrict__ nmax,
                       float* __restrict__ nsum, int NB) {
  int i = blockIdx.x * blockDim.x + threadIdx.x;
  if (i < NB) { pred[i] = 0.f; nmax[i] = -FLT_MAX; nsum[i] = 0.f; }
}

// per-(node,batch) attention scores: s_src/s_dst = feature[n,b,:] . w_{src,dst}
__global__ void k_scores(const float* __restrict__ feat,
                         const float* __restrict__ wsrc, const float* __restrict__ wdst,
                         float* __restrict__ s_src, float* __restrict__ s_dst, int NB) {
  int i = blockIdx.x * blockDim.x + threadIdx.x;
  if (i >= NB) return;
  const float4* p  = (const float4*)(feat + (long long)i * TT);
  const float4* ws = (const float4*)wsrc;
  const float4* wd = (const float4*)wdst;
  float a = 0.f, d = 0.f;
#pragma unroll
  for (int j = 0; j < 8; ++j) {
    float4 f = p[j], s4 = ws[j], d4 = wd[j];
    a += f.x * s4.x + f.y * s4.y + f.z * s4.z + f.w * s4.w;
    d += f.x * d4.x + f.y * d4.y + f.z * d4.z + f.w * d4.w;
  }
  s_src[i] = a;
  s_dst[i] = d;
}

// edge logits + segment max
__global__ void k_logits(const int* __restrict__ src, const int* __restrict__ dst,
                         const float* __restrict__ s_src, const float* __restrict__ s_dst,
                         float* __restrict__ e_att, float* __restrict__ nmax, int EB) {
  int i = blockIdx.x * blockDim.x + threadIdx.x;
  if (i >= EB) return;
  int e = i >> 6, b = i & 63;
  float x = s_src[(long long)src[e] * BC + b] + s_dst[(long long)dst[e] * BC + b];
  x = (x > 0.f) ? x : 0.01f * x;   // leaky_relu, default slope 0.01
  e_att[i] = x;
  atomicMaxFloat(nmax + (long long)dst[e] * BC + b, x);
}

// exp(e - max) + segment sum (e_att rewritten in place with exp)
__global__ void k_exp(const int* __restrict__ dst, const float* __restrict__ nmax,
                      float* __restrict__ e_att, float* __restrict__ nsum, int EB) {
  int i = blockIdx.x * blockDim.x + threadIdx.x;
  if (i >= EB) return;
  int e = i >> 6, b = i & 63;
  long long d = (long long)dst[e] * BC + b;
  float ex = expf(e_att[i] - nmax[d]);
  e_att[i] = ex;
  atomicAdd(nsum + d, ex);
}

// Main fused kernel: velocity NAM (WMMA) + travel time + geometric diffusion
// + attention-normalized scatter-add into pred.
__global__ void __launch_bounds__(256)
k_main(const float* __restrict__ up, const float* __restrict__ down,
       const float* __restrict__ feat, const float* __restrict__ dist,
       const int* __restrict__ src, const int* __restrict__ dst,
       const float* __restrict__ alpha,
       const float* __restrict__ W11, const float* __restrict__ b11,
       const float* __restrict__ g11, const float* __restrict__ be11,
       const float* __restrict__ W12, const float* __restrict__ b12,
       const float* __restrict__ W21, const float* __restrict__ b21,
       const float* __restrict__ g21, const float* __restrict__ be21,
       const float* __restrict__ W22, const float* __restrict__ b22,
       const float* __restrict__ W3, const float* __restrict__ b3,
       const float* __restrict__ e_exp, const float* __restrict__ nsum,
       float* __restrict__ pred, int E)
{
  __shared__ float vlds[8 * 32];          // 8 waves x 32 velocities
  const int lane  = threadIdx.x & 31;
  const int wave  = threadIdx.x >> 5;
  const int gwave = blockIdx.x * (blockDim.x >> 5) + wave;
  const int nwave = gridDim.x * (blockDim.x >> 5);

  const int l15 = lane & 15;
  const int k0  = (lane < 16) ? 0 : 8;    // K offset for this lane half (16-bit A/B layout)

  // B fragments: column h of Wt == row h of W (B = W^T), same index pattern as A rows.
  const v16h Bu0 = frag_load_f32(W11 + l15 * TT, k0);
  const v16h Bu1 = frag_load_f32(W11 + (16 + l15) * TT, k0);
  const v16h Bd0 = frag_load_f32(W21 + l15 * TT, k0);
  const v16h Bd1 = frag_load_f32(W21 + (16 + l15) * TT, k0);

  const float bu0 = b11[l15],  bu1 = b11[16 + l15];
  const float gu0 = g11[l15],  gu1 = g11[16 + l15];
  const float eu0 = be11[l15], eu1 = be11[16 + l15];
  const float wu0 = W12[l15],  wu1 = W12[16 + l15];
  const float bd0 = b21[l15],  bd1 = b21[16 + l15];
  const float gd0 = g21[l15],  gd1 = g21[16 + l15];
  const float ed0 = be21[l15], ed1 = be21[16 + l15];
  const float wd0 = W22[l15],  wd1 = W22[16 + l15];
  const float b12s = b12[0], b22s = b22[0];
  const float w30 = W3[0], w31 = W3[1], b3s = b3[0];

  const long long RT = (long long)E * BC;
  const int nchunks  = (int)(RT >> 5);           // 32 rows per chunk

  for (int chunk = gwave; chunk < nchunks; chunk += nwave) {
    const long long rbase = (long long)chunk * 32;

    // prefetch next grid-stride chunk's flow rows (one 128B row per lane)
    // -> global_prefetch_b8, no LOADcnt cost, hides HBM latency behind WMMA work
    {
      const long long rnext = rbase + (long long)nwave * 32;
      if (rnext < RT) {
        __builtin_prefetch(up   + (rnext + lane) * TT, 0, 3);
        __builtin_prefetch(down + (rnext + lane) * TT, 0, 3);
      }
    }

#pragma unroll
    for (int t = 0; t < 2; ++t) {                // two 16-row WMMA tiles
      const long long r0 = rbase + t * 16;
      const float* uprow = up   + (r0 + l15) * TT;
      const float* dnrow = down + (r0 + l15) * TT;
      v16h Au = frag_load_f32(uprow, k0);
      v16h Ad = frag_load_f32(dnrow, k0);
      float xu[8], xd[8];
      nam_branch(Au, Bu0, Bu1, bu0, bu1, gu0, gu1, eu0, eu1, wu0, wu1, b12s, xu);
      nam_branch(Ad, Bd0, Bd1, bd0, bd1, gd0, gd1, ed0, ed1, wd0, wd1, b22s, xd);
      if (l15 == 0) {                            // leader lanes 0 & 16 own 8 rows each
        int rowoff = t * 16 + ((lane >= 16) ? 8 : 0);
#pragma unroll
        for (int v = 0; v < 8; ++v)
          vlds[wave * 32 + rowoff + v] = w30 * xu[v] + w31 * xd[v] + b3s;
      }
    }

    asm volatile("s_wait_dscnt 0" ::: "memory"); // intra-wave LDS visibility
    float vv = vlds[wave * 32 + lane];

    // one (edge,batch) row per lane
    const long long r = rbase + lane;
    const int e = (int)(r >> 6);
    const int b = (int)(r & 63);

    float T  = dist[e] / (vv + 1e-5f);
    float Ti = rintf(T * 0.1f);
    if (Ti < 0.f)  Ti = 0.f;
    if (Ti > 32.f) Ti = 31.f;
    float n = 32.f - Ti;
    if (n < 1.f) n = 1.f;
    const int ni = (int)n;

    const float F   = 1.0f / (1.0f + alpha[e] * T);
    const float omF = 1.0f - F;
    const float* fr = feat + ((long long)src[e] * BC + b) * TT;

    // hoist all 32 feature taps into registers (8x b128 loads)
    float fl[32];
#pragma unroll
    for (int j = 0; j < 8; ++j) {
      float4 f = ((const float4*)fr)[j];
      fl[4*j]   = f.x; fl[4*j+1] = f.y; fl[4*j+2] = f.z; fl[4*j+3] = f.w;
    }

    // sum_{k=0}^{n-1} F*(1-F)^(n-1-k) * feat[k]  (iterate k = n-1 .. 0)
    float acc = 0.f, p = F;
#pragma unroll
    for (int k = 31; k >= 0; --k) {
      bool m = (k < ni);
      acc = fmaf(m ? p : 0.f, fl[k], acc);
      p   = m ? p * omF : p;
    }

    const long long didx = (long long)dst[e] * BC + b;
    const float att = e_exp[r] / nsum[didx];
    atomicAdd(pred + didx, att * acc);
  }
}

// ---------------------------------------------------------------------------
// launch
// ---------------------------------------------------------------------------
extern "C" void kernel_launch(void* const* d_in, const int* in_sizes, int n_in,
                              void* d_out, int out_size, void* d_ws, size_t ws_size,
                              hipStream_t stream) {
  const float* up    = (const float*)d_in[0];
  const float* down  = (const float*)d_in[1];
  const float* feat  = (const float*)d_in[2];
  const float* dist  = (const float*)d_in[3];
  const int*   src   = (const int*)d_in[4];
  const int*   dst   = (const int*)d_in[5];
  const float* alpha = (const float*)d_in[6];
  const float* fc_w  = (const float*)d_in[7];
  const float* attnw = (const float*)d_in[8];
  const float* W11   = (const float*)d_in[9];
  const float* b11   = (const float*)d_in[10];
  const float* g11   = (const float*)d_in[11];
  const float* be11  = (const float*)d_in[12];
  const float* W12   = (const float*)d_in[13];
  const float* b12   = (const float*)d_in[14];
  const float* W21   = (const float*)d_in[15];
  const float* b21   = (const float*)d_in[16];
  const float* g21   = (const float*)d_in[17];
  const float* be21  = (const float*)d_in[18];
  const float* W22   = (const float*)d_in[19];
  const float* b22   = (const float*)d_in[20];
  const float* W3    = (const float*)d_in[21];
  const float* b3    = (const float*)d_in[22];

  const int E  = in_sizes[3];
  const int N  = in_sizes[2] / (BC * TT);
  const int NB = N * BC;
  const int EB = E * BC;

  float* ws     = (float*)d_ws;
  float* wsrc   = ws;                 // 32
  float* wdst   = ws + 32;            // 32
  float* s_src  = ws + 64;            // NB
  float* s_dst  = s_src + NB;         // NB
  float* nmax   = s_dst + NB;         // NB
  float* nsum   = nmax + NB;          // NB
  float* e_att  = nsum + NB;          // EB (logits, then exp in place)
  float* pred   = (float*)d_out;      // [N, BC]

  k_fold<<<1, 32, 0, stream>>>(fc_w, attnw, wsrc, wdst);
  k_init<<<(NB + 255) / 256, 256, 0, stream>>>(pred, nmax, nsum, NB);
  k_scores<<<(NB + 255) / 256, 256, 0, stream>>>(feat, wsrc, wdst, s_src, s_dst, NB);
  k_logits<<<(EB + 255) / 256, 256, 0, stream>>>(src, dst, s_src, s_dst, e_att, nmax, EB);
  k_exp<<<(EB + 255) / 256, 256, 0, stream>>>(dst, nmax, e_att, nsum, EB);

  k_main<<<1024, 256, 0, stream>>>(up, down, feat, dist, src, dst, alpha,
                                   W11, b11, g11, be11, W12, b12,
                                   W21, b21, g21, be21, W22, b22,
                                   W3, b3, e_att, nsum, pred, E);
}
